// SeFT_5231270166715
// MI455X (gfx1250) — matrix operand order repaired and checked by reading, not compiled, not run on hardware
//
#include <hip/hip_runtime.h>
#include <math.h>

// ---------------------------------------------------------------------------
// SeFT forward, live-dataflow only (sum(softmax)==1 identity kills the
// attention/shared/gate branches exactly). All matmuls via
// v_wmma_f32_16x16x32_f16 (fp32 accumulate). K staged through LDS with
// double-buffered async global->LDS copies (ASYNCcnt, in-order completion:
// s_wait_asynccnt ISS overlaps stage k+1 DMA with stage k WMMA).
// N tile = 32 -> 128 blocks on the big per-head GEMM so enough WGPs stream
// the 48MB weight tensor (L2-resident across graph replays, 192MB L2).
// ---------------------------------------------------------------------------

typedef __attribute__((ext_vector_type(16))) _Float16 v16h;
typedef __attribute__((ext_vector_type(8)))  float    v8f;

// Async-LDS builtin pointee types (per hipcc diagnostic: GCC-style vectors,
// addrspace(1) source / addrspace(3) dest).
typedef int v4i_ __attribute__((vector_size(16)));

#define TILE_K     32
#define LDS_STRIDE 36   // 32 + 4 pad: keeps 16B alignment, staggers LDS banks

#if defined(__gfx1250__) && \
    __has_builtin(__builtin_amdgcn_global_load_async_to_lds_b128) && \
    __has_builtin(__builtin_amdgcn_global_load_async_to_lds_b32)
#define USE_ASYNC_LDS 1
#endif

__device__ __forceinline__ void g2l_b128(const float* g, float* l) {
#ifdef USE_ASYNC_LDS
  __builtin_amdgcn_global_load_async_to_lds_b128(
      (__attribute__((address_space(1))) v4i_*)g,
      (__attribute__((address_space(3))) v4i_*)l, 0, 0);
#else
  *(float4*)l = *(const float4*)g;
#endif
}

__device__ __forceinline__ void g2l_b32(const float* g, float* l) {
#ifdef USE_ASYNC_LDS
  __builtin_amdgcn_global_load_async_to_lds_b32(
      (__attribute__((address_space(1))) int*)g,
      (__attribute__((address_space(3))) int*)l, 0, 0);
#else
  *l = *g;
#endif
}

// Wait until at most N async-LDS ops are outstanding (in-order completion:
// <=N outstanding means every older stage has fully landed in LDS).
template <int N>
__device__ __forceinline__ void g2l_wait_n() {
#ifdef USE_ASYNC_LDS
#if __has_builtin(__builtin_amdgcn_s_wait_asynccnt)
  __builtin_amdgcn_s_wait_asynccnt(N);
#else
  asm volatile("s_wait_asynccnt %0" ::"i"(N) : "memory");
#endif
#endif
}

#define CVT4(v, b, f)                                               \
  { (v)[(b) + 0] = (_Float16)(f).x; (v)[(b) + 1] = (_Float16)(f).y; \
    (v)[(b) + 2] = (_Float16)(f).z; (v)[(b) + 3] = (_Float16)(f).w; }

// C = relu(A(64xK) * B(KxN) + bias), batched via blockIdx.y.
// Block: 128 threads (4 waves), computes a 64 x NT output tile.
// Wave w owns rows [16w, 16w+16); NT/16 accumulators cover the N tile.
template <int NT>
__global__ __launch_bounds__(128)
void wmma_mlp_gemm(const float* __restrict__ A, int lda, long long strideA,
                   const float* __restrict__ B, int ldb, long long strideB,
                   const float* __restrict__ bias, int strideBias,
                   float* __restrict__ C, int ldc, long long strideC,
                   int K) {
  constexpr int NACC = NT / 16;   // accumulators per wave
  constexpr int BPT  = NT / 4;    // B b32 async issues per thread per stage
  constexpr int ISS  = 4 + BPT;   // async issues per wave per stage

  __shared__ float lds_A[2][64][LDS_STRIDE];   // [buf][row][k] (K-contiguous)
  __shared__ float lds_B[2][NT][LDS_STRIDE];   // [buf][n][k]   (transposed)

  const int tid   = threadIdx.x;
  const int lane  = tid & 31;
  const int wave  = tid >> 5;
  const int lrow  = lane & 15;  // row (A) / col (B,C) within 16-tile
  const int hi    = lane >> 4;  // lane-group select (ISA 16-bit layouts)
  const int mBase = wave * 16;
  const int nBase = blockIdx.x * NT;
  const int batch = blockIdx.y;

  A    += (long long)batch * strideA;
  B    += (long long)batch * strideB;
  bias += (long long)batch * strideBias;
  C    += (long long)batch * strideC;

  v8f acc[NACC];
  v8f zero = {};
#pragma unroll
  for (int i = 0; i < NACC; ++i) acc[i] = zero;

  // Issue one K-slab of async global->LDS copies into buffer `buf`.
  auto stage = [&](int buf, int k0) {
    // A tile (64 rows x 32 K): K-contiguous both sides -> b128 copies
#pragma unroll
    for (int i = 0; i < 4; ++i) {
      int idx4 = tid + i * 128;        // 512 float4 chunks
      int row  = idx4 >> 3;
      int kq   = (idx4 & 7) << 2;
      g2l_b128(A + (long long)row * lda + k0 + kq, &lds_A[buf][row][kq]);
    }
    // B tile (32 K x NT) transposed into LDS[n][k]: b32 copies, coalesced in n
#pragma unroll
    for (int i = 0; i < BPT; ++i) {
      int idx = tid + i * 128;
      int n   = idx % NT;
      int k   = idx / NT;
      g2l_b32(B + (long long)(k0 + k) * ldb + nBase + n, &lds_B[buf][n][k]);
    }
  };

  const int nk = K / TILE_K;
  stage(0, 0);

  for (int it = 0; it < nk; ++it) {
    const int buf = it & 1;
    if (it + 1 < nk) {
      stage(buf ^ 1, (it + 1) * TILE_K);  // prefetch next slab (other buffer)
      g2l_wait_n<ISS>();                  // current slab landed; next in flight
    } else {
      g2l_wait_n<0>();
    }
    __syncthreads();

    // ---- A fragment, 16x32 f16 (ISA 7.12.2 16-bit A layout):
    // lanes 0-15 elems 0..7 -> K 0..7, elems 8..15 -> K 16..23;
    // lanes 16-31 shifted by +8.
    const float* ap = &lds_A[buf][mBase + lrow][hi * 8];
    float4 a0 = *(const float4*)(ap + 0);
    float4 a1 = *(const float4*)(ap + 4);
    float4 a2 = *(const float4*)(ap + 16);
    float4 a3 = *(const float4*)(ap + 20);
    v16h af;
    CVT4(af, 0, a0) CVT4(af, 4, a1) CVT4(af, 8, a2) CVT4(af, 12, a3)

#pragma unroll
    for (int ns = 0; ns < NACC; ++ns) {
      // B fragment, 32x16 f16: lane holds column N=lrow,
      // elems 0..7 -> K = hi*8 + 0..7, elems 8..15 -> K = 16 + hi*8 + 0..7.
      const float* bp = &lds_B[buf][ns * 16 + lrow][hi * 8];
      float4 b0 = *(const float4*)(bp + 0);
      float4 b1 = *(const float4*)(bp + 4);
      float4 b2 = *(const float4*)(bp + 16);
      float4 b3 = *(const float4*)(bp + 20);
      v16h bf;
      CVT4(bf, 0, b0) CVT4(bf, 4, b1) CVT4(bf, 8, b2) CVT4(bf, 12, b3)
      acc[ns] = __builtin_amdgcn_wmma_f32_16x16x32_f16(
          false, af, false, bf, (short)0, acc[ns], false, false);
    }
    __syncthreads();  // all waves done reading `buf` before it is re-staged
  }

  // ---- epilogue: bias + relu, C/D layout: VGPR r -> M = hi*8 + r, N = lrow
#pragma unroll
  for (int ns = 0; ns < NACC; ++ns) {
    int col  = nBase + ns * 16 + lrow;
    float bv = bias[col];
#pragma unroll
    for (int r = 0; r < 8; ++r) {
      int row = mBase + hi * 8 + r;
      float v = acc[ns][r] + bv;
      C[(long long)row * ldc + col] = v > 0.f ? v : 0.f;
    }
  }
}

// out[b] = sigmoid(dot(Y2[b,:], f3_w) + f3_b)
__global__ void final_sigmoid(const float* __restrict__ Y,
                              const float* __restrict__ w,
                              const float* __restrict__ b,
                              float* __restrict__ out) {
  int t = threadIdx.x;  // 64 threads, one per batch row
  float s = b[0];
#pragma unroll
  for (int j = 0; j < 64; ++j) s = fmaf(Y[t * 64 + j], w[j], s);
  out[t] = 1.0f / (1.0f + __expf(-s));
}

extern "C" void kernel_launch(void* const* d_in, const int* in_sizes, int n_in,
                              void* d_out, int out_size, void* d_ws, size_t ws_size,
                              hipStream_t stream) {
  // setup_inputs order:
  // 0 x, 1 n, 2 query, 3 sh1_w, 4 sh1_b, 5 sh2_w, 6 sh2_b, 7 sg1_w, 8 sg1_b,
  // 9 sg2_w, 10 sg2_b, 11 attn_w, 12 ah1_w, 13 ah1_b, 14 ah2_w, 15 ah2_b,
  // 16 f1_w, 17 f1_b, 18 f2_w, 19 f2_b, 20 f3_w, 21 f3_b
  const float* x     = (const float*)d_in[0];
  const float* ah1_w = (const float*)d_in[12];
  const float* ah1_b = (const float*)d_in[13];
  const float* ah2_w = (const float*)d_in[14];
  const float* ah2_b = (const float*)d_in[15];
  const float* f1_w  = (const float*)d_in[16];
  const float* f1_b  = (const float*)d_in[17];
  const float* f2_w  = (const float*)d_in[18];
  const float* f2_b  = (const float*)d_in[19];
  const float* f3_w  = (const float*)d_in[20];
  const float* f3_b  = (const float*)d_in[21];
  float* out = (float*)d_out;

  float* H1 = (float*)d_ws;            // 4 * 64 * 1024 f32  (per-head hidden)
  float* R  = H1 + 4 * 64 * 1024;      // 64 * 512           (interleaved heads)
  float* Y1 = R  + 64 * 512;           // 64 * 256
  float* Y2 = Y1 + 64 * 256;           // 64 * 64

  dim3 blk(128, 1, 1);

  // H1[m] = relu(xf @ ah1_w[m] + ah1_b[m])        (64x3072)·(3072x1024)
  // 32 n-blocks x 4 heads = 128 blocks streaming the 48MB weight tensor.
  wmma_mlp_gemm<32><<<dim3(32, 4), blk, 0, stream>>>(
      x, 3072, 0LL,
      ah1_w, 1024, 3072LL * 1024,
      ah1_b, 1024,
      H1, 1024, 64LL * 1024, 3072);

  // R[b, m*128+d] = relu(H1[m] @ ah2_w[m] + ah2_b[m])   (64x1024)·(1024x128)
  wmma_mlp_gemm<32><<<dim3(4, 4), blk, 0, stream>>>(
      H1, 1024, 64LL * 1024,
      ah2_w, 128, 1024LL * 128,
      ah2_b, 128,
      R, 512, 128LL, 1024);

  // Y1 = relu(R @ f1_w + f1_b)                    (64x512)·(512x256)
  wmma_mlp_gemm<32><<<dim3(8, 1), blk, 0, stream>>>(
      R, 512, 0LL, f1_w, 256, 0LL, f1_b, 0, Y1, 256, 0LL, 512);

  // Y2 = relu(Y1 @ f2_w + f2_b)                   (64x256)·(256x64)
  wmma_mlp_gemm<32><<<dim3(2, 1), blk, 0, stream>>>(
      Y1, 256, 0LL, f2_w, 64, 0LL, f2_b, 0, Y2, 64, 0LL, 256);

  // out = sigmoid(Y2 @ f3_w + f3_b)
  final_sigmoid<<<1, 64, 0, stream>>>(Y2, f3_w, f3_b, out);
}
